// PiGNNLayer_36043365548233
// MI455X (gfx1250) — matrix operand order repaired and checked by reading, not compiled, not run on hardware
//
#include <hip/hip_runtime.h>

typedef __attribute__((ext_vector_type(16))) _Float16 v16h;
typedef __attribute__((ext_vector_type(8)))  _Float16 v8h;
typedef __attribute__((ext_vector_type(8)))  float    v8f;

#define NNODES 10000
#define KNBR   30
#define DDIM   128
#define NHEAD  4
#define EEDGE  (NNODES*KNBR)

// ---------------------------------------------------------------------------
// WMMA helpers
// ---------------------------------------------------------------------------
__device__ __forceinline__ v8f wmma_f16(v16h a, v16h b, v8f c) {
  // D = A(16x32 f16) * B(32x16 f16) + C(16x16 f32)
  return __builtin_amdgcn_wmma_f32_16x16x32_f16(
      /*neg_a=*/false, a, /*neg_b=*/false, b,
      /*c_mod=*/(short)0, c, /*reuse_a=*/false, /*reuse_b=*/false);
}

// A-fragment (16x32, 16-bit) from row-major LDS tile.
// ISA layout: lanes 0-15 -> M=lane, halves 0..7 = K k0..k0+7, halves 8..15 = K k0+16..k0+23
//             lanes 16-31 -> M=lane-16, halves 0..7 = K k0+8..k0+15, 8..15 = K k0+24..k0+31
__device__ __forceinline__ v16h load_a_frag(const _Float16* X, int ldx, int m0, int k0, int lane) {
  int r = lane & 15, lhi = lane >> 4;
  const _Float16* p = X + (m0 + r) * ldx + k0 + lhi * 8;
  v8h lo  = *(const v8h*)(p);
  v8h hi8 = *(const v8h*)(p + 16);
  v16h a;
#pragma unroll
  for (int i = 0; i < 8; i++) { a[i] = lo[i]; a[i + 8] = hi8[i]; }
  return a;
}

// 32 x (ntiles*16) GEMM:  Y = act(X[32 x KT*32] @ Wpacked + bias), f16 output to LDS
// act: 0 = relu, 1 = exact gelu
__device__ __forceinline__ void gemm32_f16out(const _Float16* X, int ldx, int koff, int KT,
                                              const _Float16* Wp, int ntiles,
                                              const float* bias, _Float16* Y, int ldy,
                                              int wave, int lane, int act) {
  int col = lane & 15, lhi = lane >> 4;
  for (int t = wave; t < 2 * ntiles; t += 8) {      // wave-uniform -> EXEC stays full
    int tm = t & 1, tn = t >> 1;
    v8f acc = {};
    for (int kk = 0; kk < KT; kk++) {
      v16h a = load_a_frag(X, ldx, tm * 16, koff + kk * 32, lane);
      v16h b = *(const v16h*)(Wp + ((size_t)(kk * ntiles + tn) * 32 + lane) * 16);
      acc = wmma_f16(a, b, acc);
    }
    float bb = bias[tn * 16 + col];
#pragma unroll
    for (int r = 0; r < 8; r++) {
      float v = acc[r] + bb;
      v = (act == 0) ? fmaxf(v, 0.f)
                     : 0.5f * v * (1.f + erff(v * 0.70710678118654752f));
      Y[(tm * 16 + 8 * lhi + r) * ldy + tn * 16 + col] = (_Float16)v;
    }
  }
}

// Same, but f32 output, no activation (logits / final values)
__device__ __forceinline__ void gemm32_f32out(const _Float16* X, int ldx, int KT,
                                              const _Float16* Wp, int ntiles,
                                              const float* bias, int nbias,
                                              float* Y, int ldy, int wave, int lane) {
  int col = lane & 15, lhi = lane >> 4;
  for (int t = wave; t < 2 * ntiles; t += 8) {
    int tm = t & 1, tn = t >> 1;
    v8f acc = {};
    for (int kk = 0; kk < KT; kk++) {
      v16h a = load_a_frag(X, ldx, tm * 16, kk * 32, lane);
      v16h b = *(const v16h*)(Wp + ((size_t)(kk * ntiles + tn) * 32 + lane) * 16);
      acc = wmma_f16(a, b, acc);
    }
    int nidx = tn * 16 + col;
    float bb = (nidx < nbias) ? bias[nidx] : 0.f;
#pragma unroll
    for (int r = 0; r < 8; r++)
      Y[(tm * 16 + 8 * lhi + r) * ldy + nidx] = acc[r] + bb;
  }
}

// ---------------------------------------------------------------------------
// Weight pre-pack: f32 row-major [K x Nsrc] -> f16 fragment-major (B-matrix layout)
// frag f = ktile*(Npad/16)+ntile, within frag: [lane(0..31)][half(0..15)] contiguous
// element: K = ktile*32 + (lane>>4)*8 + (i&7) + (i>>3)*16 ; N = ntile*16 + (lane&15)
// ---------------------------------------------------------------------------
__global__ void pack_w_kernel(const float* __restrict__ W, _Float16* __restrict__ out,
                              int K, int Npad, int Nsrc) {
  int idx = blockIdx.x * blockDim.x + threadIdx.x;
  int total = K * Npad;
  if (idx >= total) return;
  int f = idx >> 9;            // /512 halves per fragment
  int within = idx & 511;
  int lane = within >> 4, i = within & 15;
  int ntiles = Npad >> 4;
  int ktile = f / ntiles, ntile = f % ntiles;
  int k = ktile * 32 + (lane >> 4) * 8 + (i & 7) + (i >> 3) * 16;
  int n = ntile * 16 + (lane & 15);
  float v = (n < Nsrc && k < K) ? W[(size_t)k * Nsrc + n] : 0.f;
  out[idx] = (_Float16)v;
}

// ---------------------------------------------------------------------------
// Fused per-node PiGNN layer: one workgroup per node (256 threads = 8 wave32)
// ---------------------------------------------------------------------------
__global__ __launch_bounds__(256)
void pignn_fused_kernel(const float* __restrict__ h, const float* __restrict__ e,
                        const int* __restrict__ eidx,
                        const float* __restrict__ att_b1, const float* __restrict__ att_b2,
                        const float* __restrict__ att_b3,
                        const float* __restrict__ node_b1, const float* __restrict__ node_b2,
                        const float* __restrict__ node_b3,
                        const float* __restrict__ to_h_w,
                        const _Float16* __restrict__ w_att1, const _Float16* __restrict__ w_att2,
                        const _Float16* __restrict__ w_att3,
                        const _Float16* __restrict__ w_node1, const _Float16* __restrict__ w_node2,
                        const _Float16* __restrict__ w_node3,
                        float* __restrict__ out) {
  __shared__ __align__(32) _Float16 Xin[32 * 392];   // [hi | e | hj], 32 rows (30 real + 2 pad)
  __shared__ __align__(32) _Float16 actA[32 * 136];
  __shared__ __align__(32) _Float16 actB[32 * 136];
  __shared__ float logits[32 * 16];
  __shared__ float attw[32 * 4];
  __shared__ float vout[32 * 132];
  __shared__ float outpre[128];

  const int n    = blockIdx.x;
  const int tid  = threadIdx.x;
  const int wave = tid >> 5;
  const int lane = tid & 31;
  const int* __restrict__ src = eidx;
  const int* __restrict__ dst = eidx + EEDGE;

  // ---- Stage edge inputs into LDS as f16: Xin = [hi(0:128) | e(128:256) | hj(256:384)]
#pragma unroll
  for (int i = 0; i < 4; i++) {
    int r = wave * 4 + i;
    if (r < KNBR) {
      int eix = n * KNBR + r;
      const float* hi_p = h + (size_t)src[eix] * DDIM;
      const float* e_p  = e + (size_t)eix * DDIM;
      const float* hj_p = h + (size_t)dst[eix] * DDIM;
      for (int c = lane; c < DDIM; c += 32) {
        Xin[r * 392 + c]       = (_Float16)hi_p[c];
        Xin[r * 392 + 128 + c] = (_Float16)e_p[c];
        Xin[r * 392 + 256 + c] = (_Float16)hj_p[c];
      }
    } else {
      for (int c = lane; c < 384; c += 32) Xin[r * 392 + c] = (_Float16)0.f;
    }
  }
  __syncthreads();

  // ---- Attention MLP: relu(X@W1) -> relu(@W2) -> @W3 (4 heads, padded to 16 cols)
  gemm32_f16out(Xin, 392, 0, 12, w_att1, 8, att_b1, actA, 136, wave, lane, /*relu*/0);
  __syncthreads();
  gemm32_f16out(actA, 136, 0, 4, w_att2, 8, att_b2, actB, 136, wave, lane, /*relu*/0);
  __syncthreads();
  gemm32_f32out(actB, 136, 4, w_att3, 1, att_b3, NHEAD, logits, 16, wave, lane);
  __syncthreads();

  // ---- Softmax over K=30 neighbors per head (scaled by 1/sqrt(DH)=1/sqrt(32))
  if (tid < NHEAD) {
    const float scale = 0.17677669529663687f;
    float mx = -3.4e38f;
    for (int k = 0; k < KNBR; k++) {
      float v = logits[k * 16 + tid] * scale;
      mx = fmaxf(mx, v);
    }
    float s = 0.f;
    for (int k = 0; k < KNBR; k++) {
      float v = expf(logits[k * 16 + tid] * scale - mx);
      attw[k * 4 + tid] = v;
      s += v;
    }
    float inv = 1.f / s;
    for (int k = 0; k < KNBR; k++) attw[k * 4 + tid] *= inv;
  }
  __syncthreads();

  // ---- Value MLP on [e|hj] (cols 128..383): gelu -> gelu -> linear
  gemm32_f16out(Xin, 392, 128, 8, w_node1, 8, node_b1, actA, 136, wave, lane, /*gelu*/1);
  __syncthreads();
  gemm32_f16out(actA, 136, 0, 4, w_node2, 8, node_b2, actB, 136, wave, lane, /*gelu*/1);
  __syncthreads();
  gemm32_f32out(actB, 136, 4, w_node3, 8, node_b3, DDIM, vout, 132, wave, lane);
  __syncthreads();

  // ---- Weighted aggregation over neighbors: out[h,d] = sum_k att[k,h] * v[k, h*32+d]
  if (tid < DDIM) {
    int hh = tid >> 5;
    float acc = 0.f;
    for (int k = 0; k < KNBR; k++) acc += attw[k * 4 + hh] * vout[k * 132 + tid];
    outpre[tid] = acc;
  }
  __syncthreads();

  // ---- Final projection: out[n] = outpre @ to_h_w  (to_h_w is L2-resident f32)
  if (tid < DDIM) {
    float acc = 0.f;
    for (int d2 = 0; d2 < DDIM; d2++) acc += outpre[d2] * to_h_w[d2 * DDIM + tid];
    out[(size_t)n * DDIM + tid] = acc;
  }
}

// ---------------------------------------------------------------------------
extern "C" void kernel_launch(void* const* d_in, const int* in_sizes, int n_in,
                              void* d_out, int out_size, void* d_ws, size_t ws_size,
                              hipStream_t stream) {
  (void)in_sizes; (void)n_in; (void)out_size; (void)ws_size;
  const float* h       = (const float*)d_in[0];
  const float* e       = (const float*)d_in[1];
  const int*   eidx    = (const int*)d_in[2];
  const float* att_w1  = (const float*)d_in[3];
  const float* att_b1  = (const float*)d_in[4];
  const float* att_w2  = (const float*)d_in[5];
  const float* att_b2  = (const float*)d_in[6];
  const float* att_w3  = (const float*)d_in[7];
  const float* att_b3  = (const float*)d_in[8];
  const float* node_w1 = (const float*)d_in[9];
  const float* node_b1 = (const float*)d_in[10];
  const float* node_w2 = (const float*)d_in[11];
  const float* node_b2 = (const float*)d_in[12];
  const float* node_w3 = (const float*)d_in[13];
  const float* node_b3 = (const float*)d_in[14];
  const float* to_h_w  = (const float*)d_in[15];

  // Workspace: f16 fragment-major packed weights (~260 KB, stays in L2)
  _Float16* wp       = (_Float16*)d_ws;
  _Float16* w_att1   = wp;             // 384x128 -> 49152 halves
  _Float16* w_att2   = wp + 49152;     // 128x128 -> 16384
  _Float16* w_att3   = wp + 65536;     // 128x16 (pad from 4) -> 2048
  _Float16* w_node1  = wp + 67584;     // 256x128 -> 32768
  _Float16* w_node2  = wp + 100352;    // 128x128 -> 16384
  _Float16* w_node3  = wp + 116736;    // 128x128 -> 16384

  auto pack = [&](const float* W, _Float16* o, int K, int Npad, int Nsrc) {
    int total = K * Npad;
    pack_w_kernel<<<(total + 255) / 256, 256, 0, stream>>>(W, o, K, Npad, Nsrc);
  };
  pack(att_w1,  w_att1,  384, 128, 128);
  pack(att_w2,  w_att2,  128, 128, 128);
  pack(att_w3,  w_att3,  128, 16,  4);
  pack(node_w1, w_node1, 256, 128, 128);
  pack(node_w2, w_node2, 128, 128, 128);
  pack(node_w3, w_node3, 128, 128, 128);

  pignn_fused_kernel<<<NNODES, 256, 0, stream>>>(
      h, e, eidx, att_b1, att_b2, att_b3, node_b1, node_b2, node_b3, to_h_w,
      w_att1, w_att2, w_att3, w_node1, w_node2, w_node3, (float*)d_out);
}
